// EncoderImage_3289944949024
// MI455X (gfx1250) — matrix operand order repaired and checked by reading, not compiled, not run on hardware
//
#include <hip/hip_runtime.h>
#include <math.h>
#include <stdint.h>

// ---------------------------------------------------------------------------
// MI455X / gfx1250 implementation.
// All five GEMMs run on v_wmma_f32_16x16x32_bf16 (f32 accumulate), with
// GLOBAL_LOAD_ASYNC_TO_LDS_B128 double-buffered tile staging (ASYNCcnt).
// ---------------------------------------------------------------------------

typedef __attribute__((ext_vector_type(16))) __bf16        v16bf;
typedef __attribute__((ext_vector_type(8)))  float         v8f;
typedef __attribute__((ext_vector_type(8)))  unsigned int  v8u;
typedef __attribute__((ext_vector_type(4)))  int           i32x4;

static constexpr int NB   = 256;   // batch
static constexpr int NK   = 36;    // regions
static constexpr int ND   = 2048;  // img dim
static constexpr int NE   = 1024;  // embed
static constexpr int NP   = 5;     // top-P
static constexpr int NROW = NB * NK;  // 9216
static constexpr int DPAD = 2080;     // 2053 padded to mult of 32

// LDS leading dims (bf16 units). Row strides are 80B / 272B: 16B-aligned for
// async b128 deposits, and odd-multiple-of-4-bank strides for low-conflict
// fragment reads.
static constexpr int SA = 40;
static constexpr int SB = 136;

// ---------------------------------------------------------------------------
// Async global->LDS copy (16B per lane), tracked by ASYNCcnt.
// Builtin signature (from hipcc diagnostic): (int4 AS1*, int4 AS3*, imm, imm).
// ---------------------------------------------------------------------------
__device__ inline void async_copy_b128(void* lds, const void* g) {
#if __has_builtin(__builtin_amdgcn_global_load_async_to_lds_b128)
  __builtin_amdgcn_global_load_async_to_lds_b128(
      (__attribute__((address_space(1))) i32x4*)(uintptr_t)g,
      (__attribute__((address_space(3))) i32x4*)(unsigned)(uintptr_t)lds,
      0, 0);
#else
  unsigned long long ga = (unsigned long long)(uintptr_t)g;
  unsigned la = (unsigned)(uintptr_t)lds;
  asm volatile("global_load_async_to_lds_b128 %0, %1, off"
               :: "v"(la), "v"(ga)
               : "memory");
#endif
}

__device__ inline void wait_async_zero() {
#if __has_builtin(__builtin_amdgcn_s_wait_asynccnt)
  __builtin_amdgcn_s_wait_asynccnt(0);
#else
  asm volatile("s_wait_asynccnt 0x0" ::: "memory");
#endif
}

// ---------------------------------------------------------------------------
// LDS fragment loaders, matching CDNA5 ISA 7.12.2 VGPR layouts (wave32).
// A 16x32 bf16 tile:  lane = M + 16*(k-group), vgpr j holds K pair
//   k = (j>>2)*16 + (lane>>4)*8 + (j&3)*2
// B 32x16 bf16 tile:  lane = N + 16*(k-group), element s holds
//   k = (s>>3)*16 + (lane>>4)*8 + (s&7)
// C/D f32 16x16:      vgpr r -> M = r + 8*(lane>>4), N = lane&15
// ---------------------------------------------------------------------------
__device__ inline v16bf load_a_frag(const __bf16* As, int m0, int lane) {
  const int M = m0 + (lane & 15);
  const int h = lane >> 4;
  v8u u;
#pragma unroll
  for (int j = 0; j < 8; ++j) {
    const int k = ((j >> 2) << 4) + (h << 3) + ((j & 3) << 1);
    u[j] = *reinterpret_cast<const unsigned int*>(As + M * SA + k);
  }
  return __builtin_bit_cast(v16bf, u);
}

__device__ inline v16bf load_b_frag(const __bf16* Bs, int n0, int lane) {
  const int N = n0 + (lane & 15);
  const int h = lane >> 4;
  v16bf v;
#pragma unroll
  for (int s = 0; s < 16; ++s) {
    const int k = ((s >> 3) << 4) + (h << 3) + (s & 7);
    v[s] = Bs[k * SB + N];
  }
  return v;
}

// ---------------------------------------------------------------------------
// Tiled GEMM:  C[M,Nc] = act(A[M,Kd](bf16) @ W[Kd,Nc](bf16) + bias)
// Block tile 128x128, BK=32, 256 threads = 8 waves in 4(M) x 2(N);
// each wave owns 32x64 -> 8 WMMA accumulators, 8 WMMAs / K-step.
// Double-buffered LDS fed by async b128 copies; ONE barrier per K-step.
// ---------------------------------------------------------------------------
__global__ __launch_bounds__(256)
void gemm_bf16_wmma_kernel(const __bf16* __restrict__ A, int lda,
                           const __bf16* __restrict__ W, int ldw,
                           const float* __restrict__ bias,
                           float* __restrict__ Cf, __bf16* __restrict__ Cb,
                           int Kd, int Nc, int do_relu) {
  __shared__ __attribute__((aligned(16))) __bf16 As[2][128 * SA];
  __shared__ __attribute__((aligned(16))) __bf16 Bs[2][32 * SB];

  const int tid  = threadIdx.x;
  const int lane = tid & 31;
  const int wave = tid >> 5;
  const int wm   = wave & 3;   // wave M index (32 rows each)
  const int wn   = wave >> 2;  // wave N index (64 cols each)
  const int blockRow = blockIdx.y * 128;
  const int blockCol = blockIdx.x * 128;

  // staging decomposition: A 128x32 in two 64-row halves, B 32x128 in two
  // 16-row halves; each thread moves 4 x 16B per K-step.
  const int arow = tid >> 2;          // 0..63
  const int acol = (tid & 3) << 3;    // 0,8,16,24
  const int brow = tid >> 4;          // 0..15
  const int bcol = (tid & 15) << 3;   // 0..120

  const __bf16* gA0 = A + (size_t)(blockRow + arow) * lda + acol;
  const __bf16* gA1 = gA0 + (size_t)64 * lda;
  const __bf16* gB0 = W + (size_t)brow * ldw + blockCol + bcol;
  const __bf16* gB1 = gB0 + (size_t)16 * ldw;

  auto issue_tile = [&](int buf) {
    async_copy_b128(&As[buf][arow * SA + acol], gA0);
    async_copy_b128(&As[buf][(arow + 64) * SA + acol], gA1);
    async_copy_b128(&Bs[buf][brow * SB + bcol], gB0);
    async_copy_b128(&Bs[buf][(brow + 16) * SB + bcol], gB1);
    gA0 += 32;
    gA1 += 32;
    gB0 += (size_t)32 * ldw;
    gB1 += (size_t)32 * ldw;
  };

  v8f acc[2][4];
#pragma unroll
  for (int mi = 0; mi < 2; ++mi)
#pragma unroll
    for (int ni = 0; ni < 4; ++ni)
#pragma unroll
      for (int r = 0; r < 8; ++r) acc[mi][ni][r] = 0.0f;

  const int ksteps = Kd >> 5;
  issue_tile(0);
  for (int kt = 0; kt < ksteps; ++kt) {
    const int cur = kt & 1;
    wait_async_zero();   // my async deposits for buf `cur` have landed
    __syncthreads();     // everyone's have; everyone done reading buf cur^1
    if (kt + 1 < ksteps) issue_tile(cur ^ 1);

    const __bf16* as = As[cur];
    const __bf16* bs = Bs[cur];
    const v16bf a0 = load_a_frag(as, wm * 32, lane);
    const v16bf a1 = load_a_frag(as, wm * 32 + 16, lane);
    v16bf b[4];
#pragma unroll
    for (int ni = 0; ni < 4; ++ni) b[ni] = load_b_frag(bs, wn * 64 + ni * 16, lane);

#pragma unroll
    for (int ni = 0; ni < 4; ++ni) {
      acc[0][ni] = __builtin_amdgcn_wmma_f32_16x16x32_bf16(
          false, a0, false, b[ni], (short)0, acc[0][ni], false, false);
      acc[1][ni] = __builtin_amdgcn_wmma_f32_16x16x32_bf16(
          false, a1, false, b[ni], (short)0, acc[1][ni], false, false);
    }
  }

  // Epilogue: bias + optional relu, store f32 or bf16.
  const int nl = lane & 15;
  const int mh = (lane >> 4) << 3;
#pragma unroll
  for (int mi = 0; mi < 2; ++mi)
#pragma unroll
    for (int ni = 0; ni < 4; ++ni) {
      const int col = blockCol + wn * 64 + ni * 16 + nl;
      const float bv = bias ? bias[col] : 0.0f;
#pragma unroll
      for (int r = 0; r < 8; ++r) {
        const int row = blockRow + wm * 32 + mi * 16 + mh + r;
        float x = acc[mi][ni][r] + bv;
        if (do_relu) x = fmaxf(x, 0.0f);
        const size_t idx = (size_t)row * Nc + col;
        if (Cb) Cb[idx] = (__bf16)x;
        else    Cf[idx] = x;
      }
    }
}

// ---------------------------------------------------------------------------
// Elementwise / reduction kernels
// ---------------------------------------------------------------------------
__device__ inline float block_reduce_sum(float v, float* sred) {
  const int tid = threadIdx.x;
  sred[tid] = v;
  __syncthreads();
  for (int s = 128; s > 0; s >>= 1) {
    if (tid < s) sred[tid] += sred[tid + s];
    __syncthreads();
  }
  return sred[0];
}

// f32 -> bf16 weight convert with zero row padding (R..Rp-1 zeroed)
__global__ void convert_pad_kernel(const float* __restrict__ src,
                                   __bf16* __restrict__ dst, int R, int C,
                                   long total) {
  const long i = (long)blockIdx.x * blockDim.x + threadIdx.x;
  if (i >= total) return;
  const long r = i / C;
  const long c = i - r * C;
  const float v = (r < R) ? src[r * (long)C + c] : 0.0f;
  dst[i] = (__bf16)v;
}

// Build xb (9216 x 2080 bf16) = [images | 0.1*(bbox,area) | zeros]
__global__ void build_x_kernel(const float* __restrict__ images,
                               const float* __restrict__ bboxes,
                               __bf16* __restrict__ xb) {
  const long i = (long)blockIdx.x * blockDim.x + threadIdx.x;
  const long total = (long)NROW * DPAD;
  if (i >= total) return;
  const long row = i / DPAD;
  const int col = (int)(i - row * DPAD);
  float v;
  if (col < ND) {
    v = images[row * ND + col];
  } else if (col < ND + 5) {
    const float* bb = bboxes + row * 4;
    const int c = col - ND;
    v = (c < 4) ? bb[c] * 0.1f
                : (bb[2] - bb[0]) * (bb[3] - bb[1]) * 0.1f;
  } else {
    v = 0.0f;
  }
  xb[i] = (__bf16)v;
}

// gate layer 2: m[row] = sigmoid(dot(relu_h1[row], gate_w2) + b2)
__global__ __launch_bounds__(256)
void gate2_kernel(const __bf16* __restrict__ H1, const float* __restrict__ w2,
                  const float* __restrict__ b2, float* __restrict__ m) {
  __shared__ float sred[256];
  const int row = blockIdx.x;
  const __bf16* h = H1 + (size_t)row * ND;
  float p = 0.0f;
  for (int j = threadIdx.x; j < ND; j += 256) p += (float)h[j] * w2[j];
  const float s = block_reduce_sum(p, sred);
  if (threadIdx.x == 0) m[row] = 1.0f / (1.0f + __expf(-(s + b2[0])));
}

// top-5 relation gather (values are 0/1; jax top_k is stable -> first 5 ones,
// remaining slots fall back to self index), weighted sum, l2norm, + images.
__global__ __launch_bounds__(256)
void gather_norm_kernel(const float* __restrict__ img_range,
                        const float* __restrict__ m,
                        const float* __restrict__ V,
                        const float* __restrict__ images,
                        __bf16* __restrict__ imgsb) {
  __shared__ float sred[256];
  __shared__ int ids[NP];
  __shared__ float mw[NP];
  const int row = blockIdx.x;  // b*36 + k
  const int b = row / NK;
  const int k = row - b * NK;
  if (threadIdx.x == 0) {
    const float* rr = img_range + (size_t)row * NK;
    int cnt = 0;
    for (int j = 0; j < NK && cnt < NP; ++j)
      if (rr[j] == 1.0f) ids[cnt++] = j;
    for (; cnt < NP; ++cnt) ids[cnt] = k;
    for (int p = 0; p < NP; ++p) mw[p] = m[b * NK + ids[p]];
  }
  __syncthreads();

  float o[8];
  float ss = 0.0f;
  const size_t vbase = (size_t)b * NK * ND;
#pragma unroll
  for (int q = 0; q < 8; ++q) {
    const int d = threadIdx.x + q * 256;
    float s = 0.0f;
#pragma unroll
    for (int p = 0; p < NP; ++p)
      s += mw[p] * V[vbase + (size_t)ids[p] * ND + d];
    o[q] = s;
    ss += s * s;
  }
  const float S = block_reduce_sum(ss, sred);
  const float inv = 1.0f / (sqrtf(S) + 1e-8f);
  const float* img = images + (size_t)row * ND;
  __bf16* dst = imgsb + (size_t)row * ND;
#pragma unroll
  for (int q = 0; q < 8; ++q) {
    const int d = threadIdx.x + q * 256;
    dst[d] = (__bf16)(img[d] + o[q] * inv);
  }
}

// in-place row l2norm of the final f32 output
__global__ __launch_bounds__(256)
void l2norm_rows_kernel(float* __restrict__ out, int C) {
  __shared__ float sred[256];
  const int row = blockIdx.x;
  float* r = out + (size_t)row * C;
  float ss = 0.0f;
  for (int j = threadIdx.x; j < C; j += 256) {
    const float x = r[j];
    ss += x * x;
  }
  const float S = block_reduce_sum(ss, sred);
  const float inv = 1.0f / (sqrtf(S) + 1e-8f);
  for (int j = threadIdx.x; j < C; j += 256) r[j] *= inv;
}

// ---------------------------------------------------------------------------
// Host launcher
// ---------------------------------------------------------------------------
extern "C" void kernel_launch(void* const* d_in, const int* in_sizes, int n_in,
                              void* d_out, int out_size, void* d_ws,
                              size_t ws_size, hipStream_t stream) {
  const float* images    = (const float*)d_in[0];
  const float* bboxes    = (const float*)d_in[1];
  const float* img_range = (const float*)d_in[2];
  const float* gate_w1   = (const float*)d_in[3];
  const float* gate_b1   = (const float*)d_in[4];
  const float* gate_w2   = (const float*)d_in[5];
  const float* gate_b2   = (const float*)d_in[6];
  const float* node_w1   = (const float*)d_in[7];
  const float* node_b1   = (const float*)d_in[8];
  const float* node_w2   = (const float*)d_in[9];
  const float* node_b2   = (const float*)d_in[10];
  const float* map_w1    = (const float*)d_in[11];
  const float* map_b1    = (const float*)d_in[12];
  const float* map_w2    = (const float*)d_in[13];
  const float* map_b2    = (const float*)d_in[14];

  char* ws = (char*)d_ws;
  size_t off = 0;
  auto carve = [&](size_t bytes) -> char* {
    char* p = ws + off;
    off = (off + bytes + 255) & ~(size_t)255;
    return p;
  };
  __bf16* xb   = (__bf16*)carve((size_t)NROW * DPAD * 2);
  __bf16* wg1b = (__bf16*)carve((size_t)DPAD * ND * 2);
  __bf16* wn1b = (__bf16*)carve((size_t)DPAD * ND * 2);
  __bf16* wn2b = (__bf16*)carve((size_t)ND * ND * 2);
  __bf16* wm1b = (__bf16*)carve((size_t)ND * ND * 2);
  __bf16* wm2b = (__bf16*)carve((size_t)ND * NE * 2);
  __bf16* H1   = (__bf16*)carve((size_t)NROW * ND * 2);
  float*  V    = (float*)carve((size_t)NROW * ND * 4);
  float*  mg   = (float*)carve((size_t)NROW * 4);
  __bf16* imgsb = xb;  // xb dead after both layer-1 GEMMs; reuse for imgs(bf16)

  auto nblk = [](long t) { return (unsigned)((t + 255) / 256); };

  // weight conversion (f32 -> bf16, K padded to 2080 where needed)
  convert_pad_kernel<<<nblk((long)DPAD * ND), 256, 0, stream>>>(
      gate_w1, wg1b, ND + 5, ND, (long)DPAD * ND);
  convert_pad_kernel<<<nblk((long)DPAD * ND), 256, 0, stream>>>(
      node_w1, wn1b, ND + 5, ND, (long)DPAD * ND);
  convert_pad_kernel<<<nblk((long)ND * ND), 256, 0, stream>>>(
      node_w2, wn2b, ND, ND, (long)ND * ND);
  convert_pad_kernel<<<nblk((long)ND * ND), 256, 0, stream>>>(
      map_w1, wm1b, ND, ND, (long)ND * ND);
  convert_pad_kernel<<<nblk((long)ND * NE), 256, 0, stream>>>(
      map_w2, wm2b, ND, NE, (long)ND * NE);
  build_x_kernel<<<nblk((long)NROW * DPAD), 256, 0, stream>>>(images, bboxes, xb);

  const dim3 blk(256);
  const dim3 gD(ND / 128, NROW / 128);  // 16 x 72
  const dim3 gE(NE / 128, NROW / 128);  //  8 x 72

  // gate layer 1 -> H1 (bf16, relu), then gate layer 2 -> m
  gemm_bf16_wmma_kernel<<<gD, blk, 0, stream>>>(xb, DPAD, wg1b, ND, gate_b1,
                                                nullptr, H1, DPAD, ND, 1);
  gate2_kernel<<<NROW, 256, 0, stream>>>(H1, gate_w2, gate_b2, mg);

  // node layer 1 -> H1 (reused), node layer 2 -> V (f32)
  gemm_bf16_wmma_kernel<<<gD, blk, 0, stream>>>(xb, DPAD, wn1b, ND, node_b1,
                                                nullptr, H1, DPAD, ND, 1);
  gemm_bf16_wmma_kernel<<<gD, blk, 0, stream>>>(H1, ND, wn2b, ND, node_b2,
                                                V, nullptr, ND, ND, 0);

  // relation gather + l2norm + residual -> imgs (bf16)
  gather_norm_kernel<<<NROW, 256, 0, stream>>>(img_range, mg, V, images, imgsb);

  // map layer 1 -> H1 (relu), map layer 2 -> d_out (f32), then row l2norm
  gemm_bf16_wmma_kernel<<<gD, blk, 0, stream>>>(imgsb, ND, wm1b, ND, map_b1,
                                                nullptr, H1, ND, ND, 1);
  gemm_bf16_wmma_kernel<<<gE, blk, 0, stream>>>(H1, ND, wm2b, NE, map_b2,
                                                (float*)d_out, nullptr, ND, NE, 0);
  l2norm_rows_kernel<<<NROW, 256, 0, stream>>>((float*)d_out, NE);
}
